// RevRNN_13683765805346
// MI455X (gfx1250) — matrix operand order repaired and checked by reading, not compiled, not run on hardware
//
#include <hip/hip_runtime.h>
#include <hip/hip_bf16.h>

// ---------------------------------------------------------------------------
// RevRNN on MI455X (gfx1250): 520 sequential steps, each = 2x
// (batchnorm -> mish -> [256x2048]x[2048x1024] bf16 WMMA GEMM + residual).
// Weights (8 MB bf16) are L2-resident (192 MB L2), so the recurrence is
// compute/latency bound -> maximize v_wmma_f32_16x16x32_bf16 throughput.
// ---------------------------------------------------------------------------

typedef __attribute__((ext_vector_type(16))) __bf16 v16bf;
typedef __attribute__((ext_vector_type(8)))  __bf16 v8bf;
typedef __attribute__((ext_vector_type(8)))  float  v8f;

#define BATCH 256
#define HDIM  1024
#define KDIM  2048   // H (state) + F (seq)
#define DELAY 8

union FragBF {
    v8bf  h[2];
    v16bf f;
};

static __device__ __forceinline__ unsigned short f2bf(float f) {
    unsigned int u = __float_as_uint(f);
    unsigned int r = (u + 0x7FFFu + ((u >> 16) & 1u)) >> 16;  // RNE
    return (unsigned short)r;
}

// ---------------------------------------------------------------------------
// Fused: batch stats (unbiased, over dim 0) + normalize + mish + bf16 store
// into A-buffer. Blocks [0,32): state columns. Blocks [32,64): copy x_t into
// the seq half of the A-buffer (zeros for the DELAY tail steps).
// Each block owns 32 columns across all 256 rows -> stats are block-local.
// ---------------------------------------------------------------------------
__global__ __launch_bounds__(256) void fuse_norm_mish(
    const float* __restrict__ state,          // [256 x 1024]
    const float* __restrict__ w,
    const float* __restrict__ bias,
    const float* __restrict__ x,              // fn_input [B, S, F] (may be unused)
    int t, int S,
    unsigned short* __restrict__ Abuf)        // [256 x 2048] bf16
{
    const int tx = threadIdx.x & 31;          // column within tile
    const int ty = threadIdx.x >> 5;          // row group (8 groups)

    if (blockIdx.x < 32) {
        const int col = blockIdx.x * 32 + tx;
        float s = 0.f, sq = 0.f;
        for (int r = ty; r < BATCH; r += 8) {
            float v = state[r * HDIM + col];
            s += v; sq += v * v;
        }
        __shared__ float ls[8][32], lq[8][32], lmean[32], lrstd[32];
        ls[ty][tx] = s; lq[ty][tx] = sq;
        __syncthreads();
        if (ty == 0) {
            float S1 = 0.f, S2 = 0.f;
            for (int i = 0; i < 8; ++i) { S1 += ls[i][tx]; S2 += lq[i][tx]; }
            float mean = S1 * (1.f / (float)BATCH);
            float var  = (S2 - (float)BATCH * mean * mean) * (1.f / (float)(BATCH - 1));
            lmean[tx] = mean;
            lrstd[tx] = rsqrtf(var);
        }
        __syncthreads();
        const float mean = lmean[tx], rstd = lrstd[tx];
        const float wv = w[col], bv = bias[col];
        for (int r = ty; r < BATCH; r += 8) {
            float v  = (state[r * HDIM + col] - mean) * rstd * wv + bv;
            float sp = (v > 20.f) ? v : log1pf(__expf(v));
            float mi = v * tanhf(sp);
            Abuf[r * KDIM + col] = f2bf(mi);
        }
    } else {
        // seq half: x_t (contiguous in F -> coalesced) or zeros past S
        const int col = (blockIdx.x - 32) * 32 + tx;
        for (int r = ty; r < BATCH; r += 8) {
            float v = (t < S) ? x[((size_t)r * (size_t)S + (size_t)t) * HDIM + col] : 0.f;
            Abuf[r * KDIM + HDIM + col] = f2bf(v);
        }
    }
}

// ---------------------------------------------------------------------------
// Out[256x1024] = A[256x2048 bf16] * W[1024x2048 bf16]^T + Cin (residual).
// 8 waves/block: 2 (M) x 4 (N); each wave: 16x64 via 4 accumulators with
// A-fragment reuse. Grid (8,4). v_wmma_f32_16x16x32_bf16, K-loop of 64 iters.
// ---------------------------------------------------------------------------
__global__ __launch_bounds__(256) void gemm_wmma_bf16(
    const unsigned short* __restrict__ A,     // [256 x 2048]
    const unsigned short* __restrict__ W,     // [1024 x 2048]
    const float* __restrict__ Cin,            // [256 x 1024] residual
    float* __restrict__ Out)                  // [256 x 1024]
{
    const int lane = threadIdx.x & 31;
    const int wave = threadIdx.x >> 5;
    const int wm = wave & 1;
    const int wn = wave >> 1;
    const int m0 = blockIdx.x * 32 + wm * 16;
    const int n0 = blockIdx.y * 256 + wn * 64;
    const int r    = lane & 15;               // A row / B col / D col
    const int half = lane >> 4;

    // Accumulators initialized from residual C (D layout: row e+8*half, col r)
    v8f acc[4];
    for (int j = 0; j < 4; ++j)
        for (int e = 0; e < 8; ++e)
            acc[j][e] = Cin[(m0 + e + 8 * half) * HDIM + n0 + j * 16 + r];

    // A fragment: lane holds row m0+r; K chunks at half*8 and 16+half*8
    const unsigned short* Arow = A + (size_t)(m0 + r) * KDIM + half * 8;
    // B fragment: lane holds col n; K chunks at half*8 and 16+half*8
    const unsigned short* Wb[4];
    for (int j = 0; j < 4; ++j)
        Wb[j] = W + (size_t)(n0 + j * 16 + r) * KDIM + half * 8;

    for (int k0 = 0; k0 < KDIM; k0 += 32) {
        FragBF a;
        a.h[0] = *(const v8bf*)(Arow + k0);
        a.h[1] = *(const v8bf*)(Arow + k0 + 16);
        #pragma unroll
        for (int j = 0; j < 4; ++j) {
            FragBF b;
            b.h[0] = *(const v8bf*)(Wb[j] + k0);
            b.h[1] = *(const v8bf*)(Wb[j] + k0 + 16);
            acc[j] = __builtin_amdgcn_wmma_f32_16x16x32_bf16(
                false, a.f, false, b.f, (short)0, acc[j], false, false);
        }
    }

    for (int j = 0; j < 4; ++j)
        for (int e = 0; e < 8; ++e)
            Out[(m0 + e + 8 * half) * HDIM + n0 + j * 16 + r] = acc[j][e];
}

// ---------------------------------------------------------------------------
__global__ __launch_bounds__(256) void cvt_weights(
    const float* __restrict__ lp0, const float* __restrict__ lp1,
    unsigned short* __restrict__ w0, unsigned short* __restrict__ w1, int n)
{
    int i = blockIdx.x * blockDim.x + threadIdx.x;
    if (i < n) { w0[i] = f2bf(lp0[i]); w1[i] = f2bf(lp1[i]); }
}

__global__ __launch_bounds__(256) void init_state(
    const float* __restrict__ h, float* __restrict__ s0, float* __restrict__ s1)
{
    int i = blockIdx.x * blockDim.x + threadIdx.x;   // 256*1024 threads
    int col = i & (HDIM - 1);
    s0[i] = h[col];
    s1[i] = h[HDIM + col];
}

__global__ __launch_bounds__(256) void write_out(
    const float* __restrict__ s0, const float* __restrict__ s1, float* __restrict__ out)
{
    int i = blockIdx.x * blockDim.x + threadIdx.x;   // 256*1024 threads
    int b = i >> 10, col = i & (HDIM - 1);
    out[b * (2 * HDIM) + col]        = s0[i];
    out[b * (2 * HDIM) + HDIM + col] = s1[i];
}

// ---------------------------------------------------------------------------
extern "C" void kernel_launch(void* const* d_in, const int* in_sizes, int n_in,
                              void* d_out, int out_size, void* d_ws, size_t ws_size,
                              hipStream_t stream) {
    const float* fn_input = (const float*)d_in[0];   // [B, S, F]
    const float* hidden   = (const float*)d_in[1];   // [2H]
    const float* lp0      = (const float*)d_in[2];   // [H, F+H]
    const float* lp1      = (const float*)d_in[3];   // [H, F+H]
    const float* bw0      = (const float*)d_in[4];
    const float* bb0      = (const float*)d_in[5];
    const float* bw1      = (const float*)d_in[6];
    // d_in[7] (bn_bias1) intentionally unused: reference passes bw1 as bias too.

    const int S     = in_sizes[0] / (BATCH * HDIM);  // 512
    const int steps = S + DELAY;                     // 520

    // Workspace layout (~13 MB)
    char* ws = (char*)d_ws;
    unsigned short* W0   = (unsigned short*)ws; ws += (size_t)HDIM * KDIM * 2;
    unsigned short* W1   = (unsigned short*)ws; ws += (size_t)HDIM * KDIM * 2;
    unsigned short* Abuf = (unsigned short*)ws; ws += (size_t)BATCH * KDIM * 2;
    float* buf[4];
    for (int i = 0; i < 4; ++i) { buf[i] = (float*)ws; ws += (size_t)BATCH * HDIM * 4; }
    float* inp0 = buf[0];   // state half 0
    float* inp1 = buf[1];   // state half 1
    float* t1   = buf[2];   // o1 scratch
    float* t0   = buf[3];   // o0 scratch

    const int WN = HDIM * KDIM;  // 2M weight elements each
    cvt_weights<<<(WN + 255) / 256, 256, 0, stream>>>(lp0, lp1, W0, W1, WN);
    init_state<<<(BATCH * HDIM) / 256, 256, 0, stream>>>(hidden, inp0, inp1);

    for (int t = 0; t < steps; ++t) {
        // o1 = calc(inp0, x_t; w0, b0, lp0) + inp1
        fuse_norm_mish<<<64, 256, 0, stream>>>(inp0, bw0, bb0, fn_input, t, S, Abuf);
        gemm_wmma_bf16<<<dim3(8, 4), 256, 0, stream>>>(Abuf, W0, inp1, t1);
        // o0 = calc(o1, x_t; w1, w1(bug), lp1) + inp0  (seq half of Abuf reused)
        fuse_norm_mish<<<32, 256, 0, stream>>>(t1, bw1, bw1, fn_input, t, S, Abuf);
        gemm_wmma_bf16<<<dim3(8, 4), 256, 0, stream>>>(Abuf, W1, inp0, t0);
        // rotate: new state = (o0, o1); retired buffers become next scratch
        float* old0 = inp0; float* old1 = inp1;
        inp0 = t0; inp1 = t1;
        t0 = old0; t1 = old1;
    }

    write_out<<<(BATCH * HDIM) / 256, 256, 0, stream>>>(inp0, inp1, (float*)d_out);
}
